// DeformableConv2d_1271310319919
// MI455X (gfx1250) — matrix operand order repaired and checked by reading, not compiled, not run on hardware
//
#include <hip/hip_runtime.h>
#include <math.h>

// ---------------- CDNA5 WMMA / TDM types ----------------
typedef __attribute__((ext_vector_type(16))) __bf16   v16bf;
typedef __attribute__((ext_vector_type(8)))  float    v8f;
typedef __attribute__((ext_vector_type(4)))  unsigned v4u;
typedef __attribute__((ext_vector_type(8)))  int      v8i;
typedef __attribute__((ext_vector_type(4)))  int      v4i;

union AFrag { unsigned u[8]; v16bf v; };
union BFrag { __bf16   b[16]; v16bf v; };

// f32 -> bf16 RNE (cold pack kernels)
__device__ __forceinline__ unsigned short f2bf(float f) {
  union { float f; unsigned u; } v; v.f = f;
  unsigned r = v.u + 0x7FFFu + ((v.u >> 16) & 1u);
  return (unsigned short)(r >> 16);
}

// Problem constants
#define BATCH 4
#define CIN   128
#define OUT   128
#define HH    128
#define WW    128
#define KK    9

// ---------------------------------------------------------------------------
// Tensor Data Mover: 1-D contiguous copy global -> LDS (D# per ISA ch.8).
// group0: count=1 | lds_addr | global_addr[56:0] | type=2
// group1: data_size=4B, tensor_dim0=tile_dim0=stride0=ndwords, 1-D tile.
// Issue from one wave only; completion tracked with TENSORcnt (in-order).
// ---------------------------------------------------------------------------
__device__ __forceinline__ void tdm_load_1d(const void* gsrc, unsigned lds_byte_off,
                                            int ndwords) {
  unsigned long long ga = (unsigned long long)(size_t)gsrc;
  v4u g0;
  g0[0] = 1u;                                   // count=1, user descriptor
  g0[1] = lds_byte_off;                         // lds_addr (bytes)
  g0[2] = (unsigned)ga;                         // global_addr[31:0]
  g0[3] = (unsigned)(ga >> 32) | (2u << 30);    // global_addr[56:32] | type=2
  v8i g1;
  g1[0] = (int)(2u << 16);                                  // data_size = 4B
  g1[1] = (int)(((unsigned)ndwords & 0xFFFFu) << 16);       // tensor_dim0[15:0]
  g1[2] = (int)(((unsigned)ndwords >> 16) & 0xFFFFu);       // tensor_dim0[31:16]
  g1[3] = (int)(((unsigned)ndwords & 0xFFFFu) << 16);       // tile_dim0
  g1[4] = 0;                                                // tile_dim1/2 = 0
  g1[5] = (int)ndwords;                                     // dim0_stride[31:0]
  g1[6] = 0;
  g1[7] = 0;
  v4i gz = {0, 0, 0, 0};
#if defined(__clang_major__) && (__clang_major__ >= 23)
  v8i z8 = {0, 0, 0, 0, 0, 0, 0, 0};
  __builtin_amdgcn_tensor_load_to_lds(g0, g1, gz, gz, z8, 0);
#else
  __builtin_amdgcn_tensor_load_to_lds(g0, g1, gz, gz, 0);
#endif
}

// ---------------------------------------------------------------------------
// Kernel 0a: pack w_reg (O,C,3,3) -> bf16 A-fragments, k-major ck = kidx*128+c.
// Packed A layout (contiguous k, B matches): lane L row M=L%16; elem e -> K = e+(L>=16?16:0).
// APACK idx = (((step*8 + t)*8 + j)*32 + lane), steps=36, o-tiles t=8.
// ---------------------------------------------------------------------------
__global__ void pack_wreg_kernel(const float* __restrict__ w_reg,
                                 unsigned* __restrict__ APACK) {
  int idx  = blockIdx.x * 256 + threadIdx.x;   // < 73728
  int lane = idx & 31;
  int j    = (idx >> 5) & 7;
  int t    = (idx >> 8) & 7;
  int step = idx >> 11;
  int sub  = step & 3;
  int kidx = step >> 2;
  int o    = t * 16 + (lane & 15);
  unsigned outv = 0;
#pragma unroll
  for (int half = 0; half < 2; ++half) {
    int e   = 2 * j + half;
    int kkA = e + ((lane >= 16) ? 16 : 0);
    int c   = sub * 32 + kkA;
    float w = w_reg[(o * CIN + c) * KK + kidx];
    outv |= (unsigned)f2bf(w) << (16 * half);
  }
  APACK[idx] = outv;
}

// ---------------------------------------------------------------------------
// Kernel 0b: pack w_off(18ch) ++ w_mod(9ch) ++ zero-pad(5) -> bf16 A-frags.
// 2 o-tiles, 36 steps -> 18432 dwords.
// ---------------------------------------------------------------------------
__global__ void pack_offmask_kernel(const float* __restrict__ w_off,
                                    const float* __restrict__ w_mod,
                                    unsigned* __restrict__ APACK_OM) {
  int idx  = blockIdx.x * 256 + threadIdx.x;   // < 18432
  int lane = idx & 31;
  int j    = (idx >> 5) & 7;
  int t    = (idx >> 8) & 1;
  int step = idx >> 9;                          // 0..35
  int sub  = step & 3;
  int kidx = step >> 2;
  int ch   = t * 16 + (lane & 15);
  unsigned outv = 0;
#pragma unroll
  for (int half = 0; half < 2; ++half) {
    int e   = 2 * j + half;
    int kkA = e + ((lane >= 16) ? 16 : 0);
    int c   = sub * 32 + kkA;
    float w = 0.f;
    if (ch < 18)      w = w_off[(ch * CIN + c) * KK + kidx];
    else if (ch < 27) w = w_mod[((ch - 18) * CIN + c) * KK + kidx];
    outv |= (unsigned)f2bf(w) << (16 * half);
  }
  APACK_OM[idx] = outv;
}

// ---------------------------------------------------------------------------
// Kernel 1: offset/mask 3x3 conv as bf16 WMMA GEMM (M=32ch pad, K=1152, N=pos).
// Block = 8 waves = one (b,ho) row; TDM double-buffers the 8 KB A slice.
// B frag = raw im2col x (tap fixed per kidx): index clamped in-bounds so the
// 16 loads are UNCONDITIONAL (batched clause), OOB zeroed by multiply.
// ---------------------------------------------------------------------------
__global__ void __launch_bounds__(256)
conv_offmask_wmma(const float* __restrict__ x,
                  const unsigned* __restrict__ APACK_OM,
                  const float* __restrict__ b_off, const float* __restrict__ b_mod,
                  float* __restrict__ OFF, float* __restrict__ MASK) {
  __shared__ __align__(16) unsigned lds_w[2][2048];     // 2 x 8 KB
  int tid = threadIdx.x, lane = tid & 31, wv = tid >> 5;
  int bid = blockIdx.x, b = bid >> 7, ho = bid & 127;
  int wo  = wv * 16 + (lane & 15);
  int sp  = ho * WW + wo;
  const float* xb = x + ((size_t)b << 21);

  unsigned ldsoff[2];
  ldsoff[0] = (unsigned)(size_t)(void*)&lds_w[0][0];
  ldsoff[1] = (unsigned)(size_t)(void*)&lds_w[1][0];

  v8f zero = {0.f, 0.f, 0.f, 0.f, 0.f, 0.f, 0.f, 0.f};
  v8f acc[2];
  acc[0] = zero; acc[1] = zero;

  if (wv == 0) tdm_load_1d(APACK_OM, ldsoff[0], 2048);

  for (int kidx = 0; kidx < 9; ++kidx) {
    if (wv == 0) {
      if (kidx < 8) {
        tdm_load_1d(APACK_OM + (kidx + 1) * 2048, ldsoff[(kidx + 1) & 1], 2048);
        __builtin_amdgcn_s_wait_tensorcnt(1);   // slice kidx landed (in-order)
      } else {
        __builtin_amdgcn_s_wait_tensorcnt(0);
      }
    }
    __syncthreads();
    const unsigned* cur = &lds_w[kidx & 1][0];

    int  yy  = ho - 1 + kidx / 3;                // tap row (uniform)
    int  xx  = wo - 1 + kidx % 3;                // tap col (per lane)
    bool ok  = (yy >= 0) & (yy < HH) & (xx >= 0) & (xx < WW);
    int  yyc = min(max(yy, 0), HH - 1);
    int  xxc = min(max(xx, 0), WW - 1);
    int  i   = yyc * WW + xxc;                   // always in-bounds
    float okf = ok ? 1.f : 0.f;

    for (int sub = 0; sub < 4; ++sub) {
      BFrag bf;
      int cb = sub * 32 + ((lane & 16) ? 16 : 0);
#pragma unroll
      for (int e = 0; e < 16; ++e) {
        const float* xp = xb + ((size_t)(cb + e) << 14);
        float s = xp[i] * okf;                   // unconditional load, masked value
        bf.b[e] = (__bf16)s;
      }
#pragma unroll
      for (int t = 0; t < 2; ++t) {
        AFrag af;
#pragma unroll
        for (int j = 0; j < 8; ++j)
          af.u[j] = cur[((sub * 2 + t) * 8 + j) * 32 + lane];
        acc[t] = __builtin_amdgcn_wmma_f32_16x16x32_bf16(
            false, af.v, false, bf.v, (short)0, acc[t], false, false);
      }
    }
    __syncthreads();
  }

  // D: VGPR r -> ch = t*16 + r + (lane>=16 ? 8 : 0); N = lane%16 = pos
#pragma unroll
  for (int t = 0; t < 2; ++t)
#pragma unroll
    for (int r = 0; r < 8; ++r) {
      int ch = t * 16 + r + ((lane & 16) ? 8 : 0);
      float v = acc[t][r];
      if (ch < 18) {
        OFF[((b * 18 + ch) << 14) + sp] = v + b_off[ch];
      } else if (ch < 27) {
        float z = v + b_mod[ch - 18];
        MASK[((b * 9 + (ch - 18)) << 14) + sp] = 2.f / (1.f + __expf(-z));
      }
    }
}

// ---------------------------------------------------------------------------
// Kernel 2: modulated deformable sampling + bf16 WMMA GEMM.
// Block = 8 waves = one (b,ho) row; wave -> 16-wide wo tile, full O=128.
// TDM double-buffers the 32 KB A slice per tap; bilinear coeffs once per
// lane per tap (k-major CK); 288 v_wmma per wave; coalesced f32 stores.
// All sample loads unconditional (clamped indices, zeroed weights).
// ---------------------------------------------------------------------------
__global__ void __launch_bounds__(256)
deform_main_kernel(const float* __restrict__ x,
                   const float* __restrict__ OFF,
                   const float* __restrict__ MASK,
                   const unsigned* __restrict__ APACK,
                   float* __restrict__ out) {
  __shared__ __align__(16) unsigned lds_a[2][8192];     // 2 x 32 KB
  int tid  = threadIdx.x, lane = tid & 31, wv = tid >> 5;
  int bid  = blockIdx.x, b = bid >> 7, ho = bid & 127;
  int wo   = wv * 16 + (lane & 15);
  int sp   = ho * WW + wo;
  const float* xb = x + ((size_t)b << 21);

  unsigned ldsoff[2];
  ldsoff[0] = (unsigned)(size_t)(void*)&lds_a[0][0];
  ldsoff[1] = (unsigned)(size_t)(void*)&lds_a[1][0];

  v8f zero = {0.f, 0.f, 0.f, 0.f, 0.f, 0.f, 0.f, 0.f};
  v8f acc[8];
#pragma unroll
  for (int t = 0; t < 8; ++t) acc[t] = zero;

  if (wv == 0) tdm_load_1d(APACK, ldsoff[0], 8192);

  for (int kidx = 0; kidx < 9; ++kidx) {
    if (wv == 0) {
      if (kidx < 8) {
        tdm_load_1d(APACK + (kidx + 1) * 8192, ldsoff[(kidx + 1) & 1], 8192);
        __builtin_amdgcn_s_wait_tensorcnt(1);
      } else {
        __builtin_amdgcn_s_wait_tensorcnt(0);
      }
    }
    __syncthreads();
    const unsigned* cur = &lds_a[kidx & 1][0];

    // ---- bilinear sampling parameters (one tap per lane) ----
    float offy = OFF[((b * 18 + 2 * kidx)     << 14) + sp];
    float offx = OFF[((b * 18 + 2 * kidx + 1) << 14) + sp];
    float m    = MASK[((b * 9 + kidx)         << 14) + sp];
    float py = (float)(ho - 1 + kidx / 3) + offy;
    float px = (float)(wo - 1 + kidx % 3) + offx;
    float fy = floorf(py), fx = floorf(px);
    int   y0 = (int)fy,    x0 = (int)fx;
    float wy1 = py - fy, wx1 = px - fx;
    float wy0 = 1.f - wy1, wx0 = 1.f - wx1;
    bool vy0 = (y0 >= 0) & (y0 < HH);
    bool vy1 = (y0 + 1 >= 0) & (y0 + 1 < HH);
    bool vx0 = (x0 >= 0) & (x0 < WW);
    bool vx1 = (x0 + 1 >= 0) & (x0 + 1 < WW);
    float f00 = (vy0 & vx0) ? wy0 * wx0 * m : 0.f;
    float f01 = (vy0 & vx1) ? wy0 * wx1 * m : 0.f;
    float f10 = (vy1 & vx0) ? wy1 * wx0 * m : 0.f;
    float f11 = (vy1 & vx1) ? wy1 * wx1 * m : 0.f;
    int y0c = min(max(y0, 0), HH - 1), y1c = min(max(y0 + 1, 0), HH - 1);
    int x0c = min(max(x0, 0), WW - 1), x1c = min(max(x0 + 1, 0), WW - 1);
    int i00 = y0c * WW + x0c, i01 = y0c * WW + x1c;
    int i10 = y1c * WW + x0c, i11 = y1c * WW + x1c;

    for (int sub = 0; sub < 4; ++sub) {
      BFrag bf;
      int cb = sub * 32 + ((lane & 16) ? 16 : 0);
#pragma unroll
      for (int e = 0; e < 16; ++e) {
        const float* xp = xb + ((size_t)(cb + e) << 14);
        float s = f00 * xp[i00] + f01 * xp[i01]
                + f10 * xp[i10] + f11 * xp[i11];
        bf.b[e] = (__bf16)s;
      }
#pragma unroll
      for (int t = 0; t < 8; ++t) {
        AFrag af;
#pragma unroll
        for (int j = 0; j < 8; ++j)
          af.u[j] = cur[((sub * 8 + t) * 8 + j) * 32 + lane];
        acc[t] = __builtin_amdgcn_wmma_f32_16x16x32_bf16(
            false, af.v, false, bf.v, (short)0, acc[t], false, false);
      }
    }
    __syncthreads();
  }

  // D layout: VGPR r -> o = t*16 + r + (lane>=16 ? 8 : 0); N = lane%16 = pos
#pragma unroll
  for (int t = 0; t < 8; ++t)
#pragma unroll
    for (int r = 0; r < 8; ++r) {
      int o = t * 16 + r + ((lane & 16) ? 8 : 0);
      out[((b * OUT + o) << 14) + sp] = acc[t][r];
    }
}

// ---------------------------------------------------------------------------
extern "C" void kernel_launch(void* const* d_in, const int* in_sizes, int n_in,
                              void* d_out, int out_size, void* d_ws, size_t ws_size,
                              hipStream_t stream) {
  const float* x     = (const float*)d_in[0];
  const float* w_off = (const float*)d_in[1];
  const float* b_off = (const float*)d_in[2];
  const float* w_mod = (const float*)d_in[3];
  const float* b_mod = (const float*)d_in[4];
  const float* w_reg = (const float*)d_in[5];
  float* out = (float*)d_out;

  // Workspace: OFF | MASK | APACK | APACK_OM  (~7.45 MB, 16B aligned)
  float*    OFF      = (float*)d_ws;                          // 4*18*16384 f32
  float*    MASK     = OFF + BATCH * 18 * HH * WW;            // 4*9*16384 f32
  unsigned* APACK    = (unsigned*)(MASK + BATCH * 9 * HH * WW); // 73728 dw
  unsigned* APACK_OM = APACK + 73728;                         // 18432 dw

  pack_wreg_kernel<<<288, 256, 0, stream>>>(w_reg, APACK);
  pack_offmask_kernel<<<72, 256, 0, stream>>>(w_off, w_mod, APACK_OM);
  conv_offmask_wmma<<<BATCH * HH, 256, 0, stream>>>(x, APACK_OM, b_off, b_mod,
                                                    OFF, MASK);
  deform_main_kernel<<<BATCH * HH, 256, 0, stream>>>(x, OFF, MASK, APACK, out);
}
